// TFAttention_77807627534899
// MI455X (gfx1250) — matrix operand order repaired and checked by reading, not compiled, not run on hardware
//
#include <hip/hip_runtime.h>

typedef __attribute__((ext_vector_type(16))) _Float16 v16h;
typedef __attribute__((ext_vector_type(8)))  _Float16 v8h;
typedef __attribute__((ext_vector_type(4)))  _Float16 v4h;
typedef __attribute__((ext_vector_type(8)))  float    v8f;
typedef __attribute__((ext_vector_type(4)))  float    v4f;

#define NB    2
#define SEQ   2048
#define NXDIM 1024
#define NHD   16
#define HD    64
#define N3    (3 * NXDIM)
#define MROWS (NB * SEQ)   // 4096

// ---------------------------------------------------------------- helpers ---

__device__ __forceinline__ v8f zero_v8f() {
  v8f z = {0.f, 0.f, 0.f, 0.f, 0.f, 0.f, 0.f, 0.f};
  return z;
}

__device__ __forceinline__ v8f wmma_f32_f16(v16h a, v16h b, v8f c) {
  // D = A(16x32 f16) * B(32x16 f16) + C(16x16 f32)
  return __builtin_amdgcn_wmma_f32_16x16x32_f16(
      /*neg_a=*/false, a, /*neg_b=*/false, b,
      /*c_mod=*/(short)0, c, /*reuse_a=*/false, /*reuse_b=*/false);
}

// A fragment (16x32, MxK), contiguous K order (matches B fragments):
//   lane<16 : M=lane,    K = 0..15
//   lane>=16: M=lane-16, K = 16..31
// rowp points at element K=0 of this lane's M row (f16, K-contiguous).
__device__ __forceinline__ v16h load_a_frag_h(const _Float16* rowp, int lane) {
  const int base = (lane & 16) ? 16 : 0;
  v8h h0 = *(const v8h*)(rowp + base);
  v8h h1 = *(const v8h*)(rowp + base + 8);
  v16h r;
#pragma unroll
  for (int i = 0; i < 8; ++i) { r[i] = h0[i]; r[8 + i] = h1[i]; }
  return r;
}

// B fragment (32x16, KxN) in this code's contiguous K order (same as A): lane&15 selects the N column,
// lanes 0-15 hold K=0..15, lanes 16-31 hold K=16..31 -> one contiguous 32B
// run when the source is K-major. Caller bakes `(lane&16)?16:0` into the ptr.

// --------------------------------------------------------- f16 staging ------

// elementwise fp32 -> f16 (x activations), 4 elements/thread, coalesced
__global__ void cvt_f32_to_f16_vec4(const float* __restrict__ src,
                                    _Float16* __restrict__ dst, size_t n4) {
  size_t i = (size_t)blockIdx.x * blockDim.x + threadIdx.x;
  if (i >= n4) return;
  v4f f = *(const v4f*)(src + i * 4);
  v4h h;
#pragma unroll
  for (int j = 0; j < 4; ++j) h[j] = (_Float16)f[j];
  *(v4h*)(dst + i * 4) = h;
}

// w[K][N] fp32 -> wT[N][K] f16 (K-major rows feed B fragments contiguously)
__global__ void cvt_transpose_f32_to_f16(const float* __restrict__ w,
                                         _Float16* __restrict__ wT,
                                         int K, int N) {
  size_t i = (size_t)blockIdx.x * blockDim.x + threadIdx.x;
  if (i >= (size_t)K * N) return;
  int n = (int)(i / K);
  int k = (int)(i % K);
  wT[i] = (_Float16)w[(size_t)k * N + n];
}

// ------------------------------------------------------------- QKV GEMM -----
// C[4096,3072] = xh[4096,1024] * w_attn + b_attn ; scatter into q/k/vT f16.
// Wave computes a 64(M) x 64(N) tile (16 WMMA / k-chunk); block = 4 waves.

__global__ __launch_bounds__(128) void qkv_gemm_wmma(
    const _Float16* __restrict__ xh, const _Float16* __restrict__ wT,
    const float* __restrict__ bias,
    _Float16* __restrict__ qh, _Float16* __restrict__ kh,
    _Float16* __restrict__ vT) {
  const int lane = threadIdx.x;
  const int wid  = blockIdx.x * 4 + threadIdx.y;
  const int tM   = (wid / (N3 / 64)) * 64;
  const int tN   = (wid % (N3 / 64)) * 64;
  const int nlo  = lane & 15;
  const int hsel = (lane & 16) ? 1 : 0;

  v8f acc[4][4];
#pragma unroll
  for (int mt = 0; mt < 4; ++mt)
#pragma unroll
    for (int nt = 0; nt < 4; ++nt) acc[mt][nt] = zero_v8f();

  const _Float16* arow[4];
  const _Float16* brow[4];
#pragma unroll
  for (int mt = 0; mt < 4; ++mt)
    arow[mt] = xh + (size_t)(tM + mt * 16 + nlo) * NXDIM;
#pragma unroll
  for (int nt = 0; nt < 4; ++nt)
    brow[nt] = wT + (size_t)(tN + nt * 16 + nlo) * NXDIM + hsel * 16;

#pragma unroll 4
  for (int kk = 0; kk < NXDIM; kk += 32) {
    v16h bf[4];
#pragma unroll
    for (int nt = 0; nt < 4; ++nt) bf[nt] = *(const v16h*)(brow[nt] + kk);
#pragma unroll
    for (int mt = 0; mt < 4; ++mt) {
      v16h af = load_a_frag_h(arow[mt] + kk, lane);
#pragma unroll
      for (int nt = 0; nt < 4; ++nt)
        acc[mt][nt] = wmma_f32_f16(af, bf[nt], acc[mt][nt]);
    }
  }

#pragma unroll
  for (int nt = 0; nt < 4; ++nt) {
    const int col3 = tN + nt * 16 + nlo;
    const float bv = bias[col3];
    const int sel  = col3 >> 10;          // 0:q 1:k 2:v
    const int col  = col3 & (NXDIM - 1);
    const int h    = col >> 6;
    const int d    = col & (HD - 1);
#pragma unroll
    for (int mt = 0; mt < 4; ++mt) {
#pragma unroll
      for (int r = 0; r < 8; ++r) {
        const int row = tM + mt * 16 + hsel * 8 + r;   // C layout row
        const int b   = row >> 11;
        const int s   = row & (SEQ - 1);
        const size_t bh = (size_t)b * NHD + h;
        const float val = acc[mt][nt][r] + bv;
        if (sel == 0)      qh[(bh * SEQ + s) * HD + d] = (_Float16)(val * 0.125f);
        else if (sel == 1) kh[(bh * SEQ + s) * HD + d] = (_Float16)val;
        else               vT[(bh * HD + d) * SEQ + s] = (_Float16)val;  // V^T
      }
    }
  }
}

// ------------------------------------------------------ flash attention -----
// One wave per (b, h, 16-row q tile). Online softmax over causal key range.

__global__ __launch_bounds__(32) void flash_attn_wmma(
    const _Float16* __restrict__ qh, const _Float16* __restrict__ kh,
    const _Float16* __restrict__ vT, _Float16* __restrict__ ah) {
  __shared__ __align__(16) _Float16 p_lds[16 * 32];

  const int lane  = threadIdx.x;
  const int qt    = blockIdx.x & (SEQ / 16 - 1);
  const int bh    = blockIdx.x >> 7;
  const int b     = bh >> 4;
  const int h     = bh & (NHD - 1);
  const int qbase = qt * 16;
  const int nlo   = lane & 15;
  const int hsel  = (lane & 16) ? 1 : 0;

  const _Float16* qp = qh + (size_t)bh * SEQ * HD;
  const _Float16* kp = kh + (size_t)bh * SEQ * HD;
  const _Float16* vp = vT + (size_t)bh * HD * SEQ;

  // Q fragments (q pre-scaled by 1/sqrt(HD) at projection time)
  const _Float16* qrow = qp + (size_t)(qbase + nlo) * HD;
  v16h aq0 = load_a_frag_h(qrow, lane);       // d 0..31
  v16h aq1 = load_a_frag_h(qrow + 32, lane);  // d 32..63

  float mrow[8], lrow[8];
  v8f o[4];
#pragma unroll
  for (int r = 0; r < 8; ++r) { mrow[r] = -1e30f; lrow[r] = 0.0f; }
#pragma unroll
  for (int j = 0; j < 4; ++j) o[j] = zero_v8f();

  const int kend = qbase + 16;                // causal: keys <= last q row
  for (int kc = 0; kc < kend; kc += 32) {
    // --- scores: S = Q * K^T, contracted over d in two K=32 WMMA steps ---
    const _Float16* krow0 = kp + (size_t)(kc + nlo) * HD + hsel * 16;
    const _Float16* krow1 = kp + (size_t)(kc + 16 + nlo) * HD + hsel * 16;
    v16h bk00 = *(const v16h*)(krow0);        // keys kc..kc+15, d 0..31
    v16h bk01 = *(const v16h*)(krow0 + 32);   // keys kc..kc+15, d 32..63
    v16h bk10 = *(const v16h*)(krow1);
    v16h bk11 = *(const v16h*)(krow1 + 32);
    v8f s0 = wmma_f32_f16(aq1, bk01, wmma_f32_f16(aq0, bk00, zero_v8f()));
    v8f s1 = wmma_f32_f16(aq1, bk11, wmma_f32_f16(aq0, bk10, zero_v8f()));

    if (kc + 32 < kend) {
      __builtin_prefetch(kp + (size_t)(kc + 32 + nlo) * HD, 0, 0);
      __builtin_prefetch(vp + (size_t)nlo * SEQ + kc + 32, 0, 0);
    }

    // --- causal mask + online softmax (rows live across 16-lane halves) ---
    const int c0 = kc + nlo;
    const int c1 = kc + 16 + nlo;
#pragma unroll
    for (int r = 0; r < 8; ++r) {
      const int row = qbase + hsel * 8 + r;
      float a0 = (c0 > row) ? -10000.0f : s0[r];   // matches w*b - 1e4*(1-b)
      float a1 = (c1 > row) ? -10000.0f : s1[r];
      float mx = fmaxf(a0, a1);
#pragma unroll
      for (int m = 8; m >= 1; m >>= 1) mx = fmaxf(mx, __shfl_xor(mx, m, 32));
      const float mn    = fmaxf(mrow[r], mx);
      const float alpha = __expf(mrow[r] - mn);
      const float p0    = __expf(a0 - mn);
      const float p1    = __expf(a1 - mn);
      float rs = p0 + p1;
#pragma unroll
      for (int m = 8; m >= 1; m >>= 1) rs += __shfl_xor(rs, m, 32);
      lrow[r] = lrow[r] * alpha + rs;
      mrow[r] = mn;
#pragma unroll
      for (int j = 0; j < 4; ++j) o[j][r] *= alpha;
      // stage P (f16) in LDS for relayout C-layout -> A-layout
      p_lds[(hsel * 8 + r) * 32 + nlo]      = (_Float16)p0;
      p_lds[(hsel * 8 + r) * 32 + 16 + nlo] = (_Float16)p1;
    }
    __syncthreads();

    // --- O += P * V  (K = 32 keys; V^T rows give contiguous B loads) ---
    v16h ap = load_a_frag_h(&p_lds[nlo * 32], lane);
#pragma unroll
    for (int j = 0; j < 4; ++j) {
      v16h bv = *(const v16h*)(vp + (size_t)(j * 16 + nlo) * SEQ + kc + hsel * 16);
      o[j] = wmma_f32_f16(ap, bv, o[j]);
    }
    __syncthreads();
  }

  // normalize and write merged-heads activations (f16) for the projection
#pragma unroll
  for (int j = 0; j < 4; ++j) {
#pragma unroll
    for (int r = 0; r < 8; ++r) {
      const int row = qbase + hsel * 8 + r;
      const int d   = j * 16 + nlo;
      const float val = o[j][r] / lrow[r];
      ah[((size_t)(b * SEQ + row)) * NXDIM + h * HD + d] = (_Float16)val;
    }
  }
}

// ------------------------------------------------------- output proj --------
// out[4096,1024] = ah * w_proj + b_proj (fp32 out). 64x64 wave tile.

__global__ __launch_bounds__(128) void proj_gemm_wmma(
    const _Float16* __restrict__ ah, const _Float16* __restrict__ wT,
    const float* __restrict__ bias, float* __restrict__ out) {
  const int lane = threadIdx.x;
  const int wid  = blockIdx.x * 4 + threadIdx.y;
  const int tM   = (wid / (NXDIM / 64)) * 64;
  const int tN   = (wid % (NXDIM / 64)) * 64;
  const int nlo  = lane & 15;
  const int hsel = (lane & 16) ? 1 : 0;

  v8f acc[4][4];
#pragma unroll
  for (int mt = 0; mt < 4; ++mt)
#pragma unroll
    for (int nt = 0; nt < 4; ++nt) acc[mt][nt] = zero_v8f();

  const _Float16* arow[4];
  const _Float16* brow[4];
#pragma unroll
  for (int mt = 0; mt < 4; ++mt)
    arow[mt] = ah + (size_t)(tM + mt * 16 + nlo) * NXDIM;
#pragma unroll
  for (int nt = 0; nt < 4; ++nt)
    brow[nt] = wT + (size_t)(tN + nt * 16 + nlo) * NXDIM + hsel * 16;

#pragma unroll 4
  for (int kk = 0; kk < NXDIM; kk += 32) {
    v16h bf[4];
#pragma unroll
    for (int nt = 0; nt < 4; ++nt) bf[nt] = *(const v16h*)(brow[nt] + kk);
#pragma unroll
    for (int mt = 0; mt < 4; ++mt) {
      v16h af = load_a_frag_h(arow[mt] + kk, lane);
#pragma unroll
      for (int nt = 0; nt < 4; ++nt)
        acc[mt][nt] = wmma_f32_f16(af, bf[nt], acc[mt][nt]);
    }
  }

#pragma unroll
  for (int nt = 0; nt < 4; ++nt) {
    const float bv = bias[tN + nt * 16 + nlo];
#pragma unroll
    for (int mt = 0; mt < 4; ++mt) {
#pragma unroll
      for (int r = 0; r < 8; ++r) {
        const int row = tM + mt * 16 + hsel * 8 + r;
        out[(size_t)row * NXDIM + tN + nt * 16 + nlo] = acc[mt][nt][r] + bv;
      }
    }
  }
}

// ----------------------------------------------------------------- launch ---

extern "C" void kernel_launch(void* const* d_in, const int* in_sizes, int n_in,
                              void* d_out, int out_size, void* d_ws, size_t ws_size,
                              hipStream_t stream) {
  (void)in_sizes; (void)n_in; (void)out_size; (void)ws_size;

  const float* x      = (const float*)d_in[0];
  const float* w_attn = (const float*)d_in[1];
  const float* b_attn = (const float*)d_in[2];
  const float* w_proj = (const float*)d_in[3];
  const float* b_proj = (const float*)d_in[4];
  float* out = (float*)d_out;

  char* ws = (char*)d_ws;
  size_t off = 0;
  _Float16* xh    = (_Float16*)(ws + off); off += (size_t)MROWS * NXDIM * 2;        // 8 MB
  _Float16* wqkvT = (_Float16*)(ws + off); off += (size_t)N3 * NXDIM * 2;           // 6 MB
  _Float16* wpT   = (_Float16*)(ws + off); off += (size_t)NXDIM * NXDIM * 2;        // 2 MB
  _Float16* qh    = (_Float16*)(ws + off); off += (size_t)NB * NHD * SEQ * HD * 2;  // 8 MB
  _Float16* kh    = (_Float16*)(ws + off); off += (size_t)NB * NHD * SEQ * HD * 2;  // 8 MB
  _Float16* vT    = (_Float16*)(ws + off); off += (size_t)NB * NHD * HD * SEQ * 2;  // 8 MB
  _Float16* ah    = (_Float16*)(ws + off); off += (size_t)MROWS * NXDIM * 2;        // 8 MB

  // activations -> f16 (coalesced), weights -> f16 K-major (transposed)
  {
    const size_t nx4 = (size_t)MROWS * NXDIM / 4;
    cvt_f32_to_f16_vec4<<<(unsigned)((nx4 + 255) / 256), 256, 0, stream>>>(x, xh, nx4);
    const size_t n1 = (size_t)NXDIM * N3;
    cvt_transpose_f32_to_f16<<<(unsigned)((n1 + 255) / 256), 256, 0, stream>>>(
        w_attn, wqkvT, NXDIM, N3);
    const size_t n2 = (size_t)NXDIM * NXDIM;
    cvt_transpose_f32_to_f16<<<(unsigned)((n2 + 255) / 256), 256, 0, stream>>>(
        w_proj, wpT, NXDIM, NXDIM);
  }

  // QKV projection (64x64 per wave, 4 waves/block)
  qkv_gemm_wmma<<<(MROWS / 64) * (N3 / 64) / 4, dim3(32, 4), 0, stream>>>(
      xh, wqkvT, b_attn, qh, kh, vT);

  // causal flash attention, one wave per 16-row q tile
  flash_attn_wmma<<<NB * NHD * (SEQ / 16), 32, 0, stream>>>(qh, kh, vT, ah);

  // output projection -> fp32
  proj_gemm_wmma<<<(MROWS / 64) * (NXDIM / 64) / 4, dim3(32, 4), 0, stream>>>(
      ah, wpT, b_proj, out);
}
